// att2seq_28020366639371
// MI455X (gfx1250) — matrix operand order repaired and checked by reading, not compiled, not run on hardware
//
#include <hip/hip_runtime.h>
#include <hip/hip_bf16.h>
#include <math.h>

typedef __bf16 bf16_t;
typedef __attribute__((ext_vector_type(16))) __bf16 v16bf;
typedef __attribute__((ext_vector_type(8)))  __bf16 v8bf;
typedef __attribute__((ext_vector_type(8)))  float   v8f;

#define HD   512
#define BB   128
#define TT   64
#define VOC  20000
#define G3   1536        // 3*HD
#define PROJ_ROWS 64     // rows per projection block (4 waves x 16)

// ---------------- WMMA fragment loaders (wave32, contiguous k order, same for A and B) ----
// A (16x32 bf16): lane ln=lane&15 -> row M; half=lane>>4.
//   vec elems 0..7  <- K = 32*ks + 16*half + [0..8)
//   vec elems 8..15 <- K = 32*ks + 16*half + 8 + [0..8)
__device__ __forceinline__ v16bf load_a_frag(const bf16_t* __restrict__ row, int ks, int half) {
  v8bf a0 = *reinterpret_cast<const v8bf*>(row + 32 * ks + 16 * half);
  v8bf a1 = *reinterpret_cast<const v8bf*>(row + 32 * ks + 16 * half + 8);
  v16bf a;
#pragma unroll
  for (int i = 0; i < 8; ++i) { a[i] = a0[i]; a[8 + i] = a1[i]; }
  return a;
}
// B (32x16 bf16): lane ln -> col N; vec elem e <- K = 32*ks + 16*half + e
// weights stored [out][in] row-major => B[k][n] = W[n][k] is contiguous in K.
__device__ __forceinline__ v16bf load_b_frag(const bf16_t* __restrict__ row, int ks, int half) {
  return *reinterpret_cast<const v16bf*>(row + 32 * ks + 16 * half);
}

__device__ __forceinline__ v8f wmma_bf16(v16bf a, v16bf b, v8f c) {
  return __builtin_amdgcn_wmma_f32_16x16x32_bf16(false, a, false, b, (short)0, c, false, false);
}

__device__ __forceinline__ float sigmoidf_(float x) { return 1.0f / (1.0f + expf(-x)); }

// ---------------- fp32 -> bf16 weight conversion ------------------------------
__global__ void cvt_bf16_kernel(const float* __restrict__ src, bf16_t* __restrict__ dst, int n) {
  int i = (blockIdx.x * blockDim.x + threadIdx.x) * 4;
  if (i < n) {
    float4 f = *reinterpret_cast<const float4*>(src + i);
    dst[i]     = (bf16_t)f.x;
    dst[i + 1] = (bf16_t)f.y;
    dst[i + 2] = (bf16_t)f.z;
    dst[i + 3] = (bf16_t)f.w;
  }
}

// ---------------- word-embedding gather -> bf16 activations -------------------
__global__ void embed_kernel(const int* __restrict__ tok, const float* __restrict__ wemb,
                             bf16_t* __restrict__ X) {
  int row = blockIdx.x;                       // row = t*B + b
  const float* src = wemb + (size_t)tok[row] * HD;
  bf16_t* dst = X + (size_t)row * HD;
  for (int d = threadIdx.x; d < HD; d += blockDim.x) dst[d] = (bf16_t)src[d];
}

// ---------------- encoder: h0 = tanh((ue+ie) @ W_h^T + b_h) (fp32, tiny) ------
__global__ void encoder_kernel(const int* __restrict__ user, const int* __restrict__ item,
                               const float* __restrict__ uemb, const float* __restrict__ iemb,
                               const float* __restrict__ Wh, const float* __restrict__ bh,
                               float* __restrict__ hf, bf16_t* __restrict__ hb) {
  int idx = blockIdx.x * blockDim.x + threadIdx.x;   // [0, B*HD)
  int b = idx / HD, h = idx % HD;
  const float* ue = uemb + (size_t)user[b] * HD;
  const float* ie = iemb + (size_t)item[b] * HD;
  const float* w  = Wh + (size_t)h * HD;
  float acc = bh[h];
  for (int d = 0; d < HD; ++d) acc += (ue[d] + ie[d]) * w[d];
  float v = tanhf(acc);
  hf[idx] = v;
  hb[idx] = (bf16_t)v;
}

// ---------------- gi_all = X @ W_ih^T + b_ih  (bf16 WMMA, fp32 out) -----------
__global__ void __launch_bounds__(32) gemm_gi_kernel(const bf16_t* __restrict__ X,
                                                     const bf16_t* __restrict__ Wih,
                                                     const float* __restrict__ bih,
                                                     float* __restrict__ gi) {
  int lane = threadIdx.x, ln = lane & 15, half = lane >> 4;
  int mbase = blockIdx.x * 16;
  int nbase = blockIdx.y * 16;
  const bf16_t* arow = X + (size_t)(mbase + ln) * HD;
  const bf16_t* brow = Wih + (size_t)(nbase + ln) * HD;
  v8f acc = {};
#pragma unroll 4
  for (int ks = 0; ks < HD / 32; ++ks) {
    v16bf a = load_a_frag(arow, ks, half);
    v16bf b = load_b_frag(brow, ks, half);
    acc = wmma_bf16(a, b, acc);
  }
  int col = nbase + ln;
  float bias = bih[col];
#pragma unroll
  for (int r = 0; r < 8; ++r) {
    int row = mbase + r + 8 * half;
    gi[(size_t)row * G3 + col] = acc[r] + bias;
  }
}

// ---------------- one GRU step: gh = h @ W_hh^T, then gates -------------------
__global__ void __launch_bounds__(32) gru_step_kernel(const bf16_t* __restrict__ hb_in,
                                                      const float* __restrict__ hf_in,
                                                      const bf16_t* __restrict__ Whh,
                                                      const float* __restrict__ bhh,
                                                      const float* __restrict__ gi_t,
                                                      float* __restrict__ hf_out,
                                                      bf16_t* __restrict__ hb_out,
                                                      bf16_t* __restrict__ outs_t) {
  int lane = threadIdx.x, ln = lane & 15, half = lane >> 4;
  int bbase = blockIdx.x * 16;   // batch tile
  int hbase = blockIdx.y * 16;   // hidden tile
  const bf16_t* arow = hb_in + (size_t)(bbase + ln) * HD;
  const bf16_t* brr = Whh + (size_t)(hbase + ln) * HD;            // r-gate rows
  const bf16_t* brz = Whh + (size_t)(hbase + ln + HD) * HD;       // z-gate rows
  const bf16_t* brn = Whh + (size_t)(hbase + ln + 2 * HD) * HD;   // n-gate rows
  v8f ar = {}, az = {}, an = {};
#pragma unroll 2
  for (int ks = 0; ks < HD / 32; ++ks) {
    v16bf a = load_a_frag(arow, ks, half);
    ar = wmma_bf16(a, load_b_frag(brr, ks, half), ar);
    az = wmma_bf16(a, load_b_frag(brz, ks, half), az);
    an = wmma_bf16(a, load_b_frag(brn, ks, half), an);
  }
  int hc = hbase + ln;
  float bhr = bhh[hc], bhz = bhh[hc + HD], bhn = bhh[hc + 2 * HD];
#pragma unroll
  for (int r = 0; r < 8; ++r) {
    int b = bbase + r + 8 * half;
    const float* girow = gi_t + (size_t)b * G3 + hc;
    float rg = sigmoidf_(girow[0]      + ar[r] + bhr);
    float zg = sigmoidf_(girow[HD]     + az[r] + bhz);
    float ng = tanhf(girow[2 * HD] + rg * (an[r] + bhn));
    float hold = hf_in[(size_t)b * HD + hc];
    float hn = (1.0f - zg) * ng + zg * hold;
    hf_out[(size_t)b * HD + hc] = hn;
    hb_out[(size_t)b * HD + hc] = (bf16_t)hn;
    outs_t[(size_t)b * HD + hc] = (bf16_t)hn;
  }
}

// ---------------- fused projection + online logsumexp + NLL -------------------
// 4 waves per block, 64 rows staged in 64KB LDS. All waves sweep the same vocab
// tiles so each B fragment is served to 4 waves from the WGP-local cache; W_out
// (20.5MB bf16) stays L2-resident and L2 traffic drops to ~2.6GB total.
__global__ void __launch_bounds__(128) proj_nll_kernel(const bf16_t* __restrict__ outs,
                                                       const bf16_t* __restrict__ Wout,
                                                       const float* __restrict__ bout,
                                                       const int* __restrict__ targets,
                                                       float* __restrict__ out_loss) {
  __shared__ __align__(16) bf16_t As[PROJ_ROWS * HD];   // 64 KB A tile, K-resident
  int tid = threadIdx.x;
  int wid = tid >> 5;                 // wave id 0..3 -> M sub-block
  int lane = tid & 31;
  int ln = lane & 15, half = lane >> 4;
  int rowbase0 = blockIdx.x * PROJ_ROWS;
  int rowbase  = rowbase0 + wid * 16;
  {
    const uint4* src = reinterpret_cast<const uint4*>(outs + (size_t)rowbase0 * HD);
    uint4* dst = reinterpret_cast<uint4*>(As);
    for (int i = tid; i < (PROJ_ROWS * HD * 2) / 16; i += 128) dst[i] = src[i];
  }
  __syncthreads();
  const bf16_t* arow = As + (size_t)(wid * 16 + ln) * HD;

  float m[8], s[8], tl[8];
  int trg[8];
#pragma unroll
  for (int r = 0; r < 8; ++r) {
    m[r] = -__builtin_inff(); s[r] = 0.0f; tl[r] = 0.0f;
    trg[r] = targets[rowbase + r + 8 * half];
  }

  for (int nt = 0; nt < VOC / 16; ++nt) {
    int nbase = nt * 16;
    const bf16_t* brow = Wout + (size_t)(nbase + ln) * HD;
    v8f acc = {};
#pragma unroll 4
    for (int ks = 0; ks < HD / 32; ++ks) {
      v16bf a = load_a_frag(arow, ks, half);
      v16bf b = load_b_frag(brow, ks, half);
      acc = wmma_bf16(a, b, acc);
    }
    int col = nbase + ln;
    float bias = bout[col];
#pragma unroll
    for (int r = 0; r < 8; ++r) {
      float val = acc[r] + bias;
      if (val > m[r]) { s[r] = s[r] * expf(m[r] - val) + 1.0f; m[r] = val; }
      else            { s[r] += expf(val - m[r]); }
      if (col == trg[r]) tl[r] += val;
    }
  }

  // merge (max,sumexp,target) across the 16 lanes of each half
#pragma unroll
  for (int r = 0; r < 8; ++r) {
#pragma unroll
    for (int off = 8; off > 0; off >>= 1) {
      float mo = __shfl_xor(m[r], off);
      float so = __shfl_xor(s[r], off);
      float to = __shfl_xor(tl[r], off);
      float mn = fmaxf(m[r], mo);
      s[r] = s[r] * expf(m[r] - mn) + so * expf(mo - mn);
      m[r] = mn;
      tl[r] += to;
    }
  }
  if (ln == 0) {
#pragma unroll
    for (int r = 0; r < 8; ++r) {
      int row = rowbase + r + 8 * half;
      float nll = (m[r] + logf(s[r])) - tl[r];
      out_loss[row] = (trg[r] != 0) ? nll : 0.0f;   // PAD == 0
    }
  }
}

// ---------------- obj = sum(loss) / max(count(tgt != 0), 1) -------------------
__global__ void obj_kernel(const float* __restrict__ loss, const int* __restrict__ targets,
                           float* __restrict__ obj) {
  __shared__ float ssum[256];
  __shared__ int scnt[256];
  int tid = threadIdx.x;
  float sum = 0.0f; int cnt = 0;
  for (int i = tid; i < TT * BB; i += 256) {
    sum += loss[i];
    cnt += (targets[i] != 0) ? 1 : 0;
  }
  ssum[tid] = sum; scnt[tid] = cnt;
  __syncthreads();
  for (int off = 128; off > 0; off >>= 1) {
    if (tid < off) { ssum[tid] += ssum[tid + off]; scnt[tid] += scnt[tid + off]; }
    __syncthreads();
  }
  if (tid == 0) obj[0] = ssum[0] / fmaxf((float)scnt[0], 1.0f);
}

// ---------------- launcher ----------------------------------------------------
extern "C" void kernel_launch(void* const* d_in, const int* in_sizes, int n_in,
                              void* d_out, int out_size, void* d_ws, size_t ws_size,
                              hipStream_t stream) {
  const int*   user = (const int*)d_in[0];
  const int*   item = (const int*)d_in[1];
  const int*   rin  = (const int*)d_in[2];
  const int*   rtg  = (const int*)d_in[3];
  const float* uemb = (const float*)d_in[4];
  const float* iemb = (const float*)d_in[5];
  const float* Wh   = (const float*)d_in[6];
  const float* bh   = (const float*)d_in[7];
  const float* wemb = (const float*)d_in[8];
  const float* Wih  = (const float*)d_in[9];
  const float* Whh  = (const float*)d_in[10];
  const float* bih  = (const float*)d_in[11];
  const float* bhh  = (const float*)d_in[12];
  const float* Wout = (const float*)d_in[13];
  const float* bout = (const float*)d_in[14];
  float* out = (float*)d_out;

  // workspace layout (all offsets 4KB-aligned)
  char* ws = (char*)d_ws;
  bf16_t* Wih_b  = (bf16_t*)(ws + 0);           // 1.5 MB
  bf16_t* Whh_b  = (bf16_t*)(ws + 1572864);     // 1.5 MB
  bf16_t* Wout_b = (bf16_t*)(ws + 3145728);     // 20.5 MB (L2-resident)
  bf16_t* Xb     = (bf16_t*)(ws + 23625728);    // 8.4 MB  embedded tokens, bf16
  float*  gi     = (float*) (ws + 32014336);    // 50.3 MB gi_all fp32
  float*  hf     = (float*) (ws + 82345984);    // 2 x 128*512 fp32 (double buffer)
  bf16_t* hb     = (bf16_t*)(ws + 82870272);    // 2 x 128*512 bf16 (double buffer)
  bf16_t* outs_b = (bf16_t*)(ws + 83132416);    // 8.4 MB  GRU outputs, bf16

  cvt_bf16_kernel<<<(G3 * HD / 4 + 255) / 256, 256, 0, stream>>>(Wih, Wih_b, G3 * HD);
  cvt_bf16_kernel<<<(G3 * HD / 4 + 255) / 256, 256, 0, stream>>>(Whh, Whh_b, G3 * HD);
  cvt_bf16_kernel<<<(VOC * HD / 4 + 255) / 256, 256, 0, stream>>>(Wout, Wout_b, VOC * HD);

  embed_kernel<<<TT * BB, 256, 0, stream>>>(rin, wemb, Xb);
  encoder_kernel<<<(BB * HD) / 256, 256, 0, stream>>>(user, item, uemb, iemb, Wh, bh, hf, hb);

  gemm_gi_kernel<<<dim3(TT * BB / 16, G3 / 16), 32, 0, stream>>>(Xb, Wih_b, bih, gi);

  for (int t = 0; t < TT; ++t) {
    const bf16_t* hbi = hb + (size_t)(t & 1) * (BB * HD);
    const float*  hfi = hf + (size_t)(t & 1) * (BB * HD);
    bf16_t* hbo = hb + (size_t)((t + 1) & 1) * (BB * HD);
    float*  hfo = hf + (size_t)((t + 1) & 1) * (BB * HD);
    gru_step_kernel<<<dim3(BB / 16, HD / 16), 32, 0, stream>>>(
        hbi, hfi, Whh_b, bhh, gi + (size_t)t * BB * G3, hfo, hbo,
        outs_b + (size_t)t * BB * HD);
  }

  proj_nll_kernel<<<TT * BB / PROJ_ROWS, 128, 0, stream>>>(outs_b, Wout_b, bout, rtg, out);
  obj_kernel<<<1, 256, 0, stream>>>(out, rtg, out + TT * BB);
}
